// LinearWithLORA_14491219657132
// MI455X (gfx1250) — matrix operand order repaired and checked by reading, not compiled, not run on hardware
//
#include <hip/hip_runtime.h>

// -------------------------------------------------------------------------
// LoRA Linear on MI455X (gfx1250), wave32 + WMMA + async global->LDS.
//   out = X @ (W^T + ALPHA * A@B) + b
// Phase 1: fold weights once into split-bf16 planes (Weff_hi, Weff_lo) in ws.
// Phase 2: memory-bound GEMM (HBM-limited ~25us; compute ~1-5us), bf16x2
//          emulated fp32: hi*hi + hi*lo + lo*hi, fp32 accumulation.
//          Double-buffered LDS; weight tiles staged with
//          GLOBAL_LOAD_ASYNC_TO_LDS_B128 (ASYNCcnt) when available.
// -------------------------------------------------------------------------

typedef __attribute__((ext_vector_type(16))) __bf16 v16bf;
typedef __attribute__((ext_vector_type(8)))  float  v8f;
typedef __attribute__((ext_vector_type(4)))  int    v4i;

#define D_IN   4096
#define D_OUT  4096
#define M_TOT  16384   // 8 * 2048
#define RANK   16
#define ALPHA  16.0f

#if defined(__HIP_DEVICE_COMPILE__) && __has_builtin(__builtin_amdgcn_global_load_async_to_lds_b128)
#define HAVE_ASYNC_LDS 1
#endif

#ifdef HAVE_ASYNC_LDS
#define GLOBAL_AS __attribute__((address_space(1)))
#define LDS_AS    __attribute__((address_space(3)))
#endif

// 16-byte global->LDS copy: async DMA path on gfx1250, sync fallback otherwise.
__device__ __forceinline__ void copy_b128(const void* g, void* l) {
#ifdef HAVE_ASYNC_LDS
  __builtin_amdgcn_global_load_async_to_lds_b128(
      (GLOBAL_AS v4i*)(void*)g, (LDS_AS v4i*)l, /*offset=*/0, /*cpol=*/0);
#else
  *reinterpret_cast<uint4*>(l) = *reinterpret_cast<const uint4*>(g);
#endif
}

__device__ __forceinline__ void wait_async_copies() {
#ifdef HAVE_ASYNC_LDS
#if __has_builtin(__builtin_amdgcn_s_wait_asynccnt)
  __builtin_amdgcn_s_wait_asynccnt(0);
#else
  asm volatile("s_wait_asynccnt 0x0" ::: "memory");
#endif
#endif
}

// ---------------- Phase 1: W_eff = W^T + ALPHA * A@B, split to bf16 hi/lo ---
// Layout of Weff planes: [o][d], d contiguous (matches W's [Dout][Din] layout)
__global__ __launch_bounds__(256)
void fold_weff(const float* __restrict__ W,   // [Dout][Din]
               const float* __restrict__ A,   // [Din][RANK]
               const float* __restrict__ B,   // [RANK][Dout]
               __bf16* __restrict__ Wh,
               __bf16* __restrict__ Wl) {
  const int tid = blockIdx.x * 256 + threadIdx.x;   // Dout * (Din/4) threads
  const int o   = tid >> 10;           // output feature
  const int d0  = (tid & 1023) << 2;   // 4 consecutive input features

  float bb[RANK];
#pragma unroll
  for (int r = 0; r < RANK; ++r) bb[r] = B[(size_t)r * D_OUT + o];

  float4 wv = *reinterpret_cast<const float4*>(W + (size_t)o * D_IN + d0);
  float w[4] = {wv.x, wv.y, wv.z, wv.w};

  struct alignas(8) BF4 { __bf16 v[4]; };
  BF4 hv, lv;
#pragma unroll
  for (int j = 0; j < 4; ++j) {
    const float* arow = A + (size_t)(d0 + j) * RANK;  // contiguous 16 floats
    float acc = 0.0f;
#pragma unroll
    for (int r = 0; r < RANK; ++r) acc += arow[r] * bb[r];
    float we  = w[j] + ALPHA * acc;
    __bf16 hi = (__bf16)we;
    hv.v[j] = hi;
    lv.v[j] = (__bf16)(we - (float)hi);
  }
  *reinterpret_cast<BF4*>(Wh + (size_t)o * D_IN + d0) = hv;
  *reinterpret_cast<BF4*>(Wl + (size_t)o * D_IN + d0) = lv;
}

// ---------------- Phase 2: GEMM, 128x128 tile / block, 8 waves -------------
#define KT    32
#define LDP   40   // row pad to 80B: spreads LDS bank accesses

__global__ __launch_bounds__(256)
void lora_gemm(const float*  __restrict__ X,     // [M][Din]
               const __bf16* __restrict__ Wh,    // [Dout][Din]
               const __bf16* __restrict__ Wl,    // [Dout][Din]
               const float*  __restrict__ bias,  // [Dout]
               float*        __restrict__ out) { // [M][Dout]
  __shared__ __bf16 sXh[2][128][LDP];
  __shared__ __bf16 sXl[2][128][LDP];
  __shared__ __bf16 sWh[2][128][LDP];
  __shared__ __bf16 sWl[2][128][LDP];

  const int t     = threadIdx.x;
  const int lane  = t & 31;
  const int wave  = t >> 5;
  const int m_blk = blockIdx.y * 128;
  const int n_blk = blockIdx.x * 128;

  // wave -> 32x64 subtile (4 row-groups x 2 col-groups of waves)
  const int rbase = (wave & 3) * 32;
  const int cbase = (wave >> 2) * 64;

  v8f acc[2][4];
#pragma unroll
  for (int i = 0; i < 2; ++i)
#pragma unroll
    for (int j = 0; j < 4; ++j) acc[i][j] = (v8f)(0.0f);

  // cooperative-load assignment: 2 threads per row/col, 16 elems each
  const int ldr = t >> 1;
  const int seg = (t & 1) * 16;
  const float*  xsrc  = X  + (size_t)(m_blk + ldr) * D_IN + seg;
  const __bf16* whsrc = Wh + (size_t)(n_blk + ldr) * D_IN + seg;
  const __bf16* wlsrc = Wl + (size_t)(n_blk + ldr) * D_IN + seg;

  // fragment addressing (per ISA VGPR layouts)
  const int half = lane >> 4;        // 0: lanes 0-15, 1: lanes 16-31
  const int lidx = lane & 15;
  const int e0a  = half * 8;         // A: K chunks {e0a..+7} and {e0a+16..+23}
  const int e0b  = half * 16;        // unused: B reads use e0a, same K chunks as A

  auto stageW = [&](int p, int k0) {
    copy_b128(whsrc + k0,     &sWh[p][ldr][seg]);
    copy_b128(whsrc + k0 + 8, &sWh[p][ldr][seg + 8]);
    copy_b128(wlsrc + k0,     &sWl[p][ldr][seg]);
    copy_b128(wlsrc + k0 + 8, &sWl[p][ldr][seg + 8]);
  };
  auto loadX = [&](int k0, float4* xr) {
#pragma unroll
    for (int q = 0; q < 4; ++q)
      xr[q] = *reinterpret_cast<const float4*>(xsrc + k0 + q * 4);
  };
  auto splitStoreX = [&](int p, const float4* xr) {
#pragma unroll
    for (int q = 0; q < 4; ++q) {
      float vv[4] = {xr[q].x, xr[q].y, xr[q].z, xr[q].w};
#pragma unroll
      for (int e = 0; e < 4; ++e) {
        __bf16 hi = (__bf16)vv[e];
        sXh[p][ldr][seg + q * 4 + e] = hi;
        sXl[p][ldr][seg + q * 4 + e] = (__bf16)(vv[e] - (float)hi);
      }
    }
  };
  auto compute = [&](int p) {
    // A fragments: 16x32 bf16, row M = lane%16, K interleaved in chunks of 8
    v16bf ah[2], al[2];
#pragma unroll
    for (int i = 0; i < 2; ++i) {
      const int r = rbase + i * 16 + lidx;
      reinterpret_cast<uint4*>(&ah[i])[0] = *reinterpret_cast<const uint4*>(&sXh[p][r][e0a]);
      reinterpret_cast<uint4*>(&ah[i])[1] = *reinterpret_cast<const uint4*>(&sXh[p][r][e0a + 16]);
      reinterpret_cast<uint4*>(&al[i])[0] = *reinterpret_cast<const uint4*>(&sXl[p][r][e0a]);
      reinterpret_cast<uint4*>(&al[i])[1] = *reinterpret_cast<const uint4*>(&sXl[p][r][e0a + 16]);
    }
    // B fragments (col N = lane%16, K interleaved in chunks of 8 like A) + 3-product WMMA
#pragma unroll
    for (int j = 0; j < 4; ++j) {
      const int c = cbase + j * 16 + lidx;
      v16bf bh, bl;
      reinterpret_cast<uint4*>(&bh)[0] = *reinterpret_cast<const uint4*>(&sWh[p][c][e0a]);
      reinterpret_cast<uint4*>(&bh)[1] = *reinterpret_cast<const uint4*>(&sWh[p][c][e0a + 16]);
      reinterpret_cast<uint4*>(&bl)[0] = *reinterpret_cast<const uint4*>(&sWl[p][c][e0a]);
      reinterpret_cast<uint4*>(&bl)[1] = *reinterpret_cast<const uint4*>(&sWl[p][c][e0a + 16]);
#pragma unroll
      for (int i = 0; i < 2; ++i) {
        acc[i][j] = __builtin_amdgcn_wmma_f32_16x16x32_bf16(
            false, ah[i], false, bh, (short)0, acc[i][j], false, false);
        acc[i][j] = __builtin_amdgcn_wmma_f32_16x16x32_bf16(
            false, ah[i], false, bl, (short)0, acc[i][j], false, false);
        acc[i][j] = __builtin_amdgcn_wmma_f32_16x16x32_bf16(
            false, al[i], false, bh, (short)0, acc[i][j], false, false);
      }
    }
  };

  // ---- prologue: stage k0 = 0 into buffer 0 ----
  stageW(0, 0);
  {
    float4 xr[4];
    loadX(0, xr);
    splitStoreX(0, xr);
  }
  wait_async_copies();
  __syncthreads();

  // ---- main loop: compute buffer p while staging p^1 ----
  int p = 0;
  for (int k0 = KT; k0 < D_IN; k0 += KT) {
    stageW(p ^ 1, k0);              // async W tiles for next step
    float4 xr[4];
    loadX(k0, xr);                  // X fp32 for next step (in flight over WMMA)
    __builtin_prefetch(xsrc + k0 + KT, 0, 1);   // global_prefetch_b8, 1 tile ahead
    compute(p);                     // 24 x v_wmma on current buffer
    splitStoreX(p ^ 1, xr);         // convert + ds_store next X tile
    wait_async_copies();
    __syncthreads();
    p ^= 1;
  }
  compute(p);

  // ---- epilogue: add bias, write fp32 (C layout: VGPR v -> M = v + 8*half) --
#pragma unroll
  for (int j = 0; j < 4; ++j) {
    const int o = n_blk + cbase + j * 16 + lidx;
    const float bv = bias[o];
#pragma unroll
    for (int i = 0; i < 2; ++i) {
      const int m0 = m_blk + rbase + i * 16 + half * 8;
#pragma unroll
      for (int v = 0; v < 8; ++v) {
        out[(size_t)(m0 + v) * D_OUT + o] = acc[i][j][v] + bv;
      }
    }
  }
}

// -------------------------------------------------------------------------
extern "C" void kernel_launch(void* const* d_in, const int* in_sizes, int n_in,
                              void* d_out, int out_size, void* d_ws, size_t ws_size,
                              hipStream_t stream) {
  (void)in_sizes; (void)n_in; (void)out_size; (void)ws_size;
  const float* X = (const float*)d_in[0];
  const float* W = (const float*)d_in[1];
  const float* b = (const float*)d_in[2];
  const float* A = (const float*)d_in[3];
  const float* B = (const float*)d_in[4];
  float* out = (float*)d_out;

  // ws: Weff hi plane (32 MB) then lo plane (32 MB)
  __bf16* Wh = (__bf16*)d_ws;
  __bf16* Wl = (__bf16*)((char*)d_ws + (size_t)D_OUT * D_IN * sizeof(__bf16));

  fold_weff<<<(D_OUT * (D_IN / 4)) / 256, 256, 0, stream>>>(W, A, B, Wh, Wl);

  dim3 grid(D_OUT / 128, M_TOT / 128);   // 32 x 128 blocks
  lora_gemm<<<grid, 256, 0, stream>>>(X, Wh, Wl, b, out);
}
